// StarEncoderLayer_3813930959519
// MI455X (gfx1250) — compile-verified, matrix operand order checked by reading, not run on hardware
//
#include <hip/hip_runtime.h>
#include <hip/hip_bf16.h>

#define NB   8
#define SEQ  4096
#define DIM  256
#define NH   8
#define DK   32
#define DV   32
#define BM   32
#define SCALE 0.17677669529663687f

typedef __attribute__((ext_vector_type(16))) __bf16 v16bf;
typedef __attribute__((ext_vector_type(8)))  float  v8f;
typedef __attribute__((ext_vector_type(4)))  unsigned int v4u;
typedef __attribute__((ext_vector_type(8)))  int  v8i_;
typedef __attribute__((ext_vector_type(4)))  int  v4i_;

// ---------------- satellite-kernel LDS layout (bytes) ----------------
#define HS_OFF   0        // (BM+2 -> padded 48) x 256 bf16
#define ES_OFF   24576    // 32 x 256 bf16
#define SS_OFF   40960    // 16 x 256 bf16 (row 0 = s, rest zero)
#define Q_OFF    49152    // 32 x 256 bf16
#define HK_OFF   65536    // 48 x 256 bf16 (34 valid)
#define HV_OFF   90112    // 48 x 256 bf16
#define EK_OFF   114688   // 32 x 256 bf16
#define EV_OFF   131072   // 32 x 256 bf16
#define SK_OFF   147456   // 16 x 256 bf16 (1 valid)
#define SV_OFF   155648   // 16 x 256 bf16
#define O_OFF    163840   // 32 x 256 bf16
#define OPF_OFF  180224   // 32 x 256 f32
#define SAT_LDS  212992

// ---------------- relay-kernel LDS layout ----------------
#define RSH_OFF  0        // 64 x 256 bf16  (TDM destination; keep at LDS offset 0)
#define RKR_OFF  32768    // 64 x 256 bf16
#define RVR_OFF  65536    // 64 x 256 bf16
#define RQR_OFF  98304    // 256 f32
#define REL_LDS  99328

// ============================================================================
// Wave-cooperative GEMM: C[MxN=rowTiles*16 x 256] = A[.. x 256] * B^T
//  A      : LDS bf16, row-major, row stride 256 elements
//  Bt     : global bf16, stored transposed (N x K row-major, K = 256)
//  Cs/Cf  : optional bf16 / f32 LDS outputs (row stride 256)
//  Each of the 8 waves grabs 16x16 tiles round-robin; K swept in steps of 32
//  via v_wmma_f32_16x16x32_bf16.  A/B fragments use contiguous k = 16*(lane/16) + s.
//  A prefetch (global_prefetch_b8) of the next tile's B row hides the L2
//  weight stream behind the WMMA chain.
// ============================================================================
__device__ __forceinline__ void gemm_bf16(
    const __bf16* As, const __bf16* __restrict__ Bt,
    __bf16* Cs, float* Cf, int rowTiles, int mValid, int wave, int lane)
{
  const int ml = lane & 15;
  const int kh = (lane >> 4) << 4;   // K half select for A/B fragments
  const int nl = lane & 15;
  const int mh = (lane >> 4) << 3;   // row half select for C fragment
  const int tot = rowTiles * 16;
  for (int t = wave; t < tot; t += 8) {
    const int row0 = (t >> 4) << 4;
    const int col0 = (t & 15) << 4;
    // prefetch next tile's B row into cache while this tile computes
    const int tn = t + 8;
    if (tn < tot)
      __builtin_prefetch(Bt + (size_t)((((tn) & 15) << 4) + nl) * DIM, 0, 1);
    v8f acc = {};
#pragma unroll
    for (int kk = 0; kk < DIM; kk += 32) {
      v16bf a  = *(const v16bf*)(As + (row0 + ml) * DIM + kk + kh);
      v16bf bm = *(const v16bf*)(Bt + (size_t)(col0 + nl) * DIM + kk + kh);
      acc = __builtin_amdgcn_wmma_f32_16x16x32_bf16(
          false, a, false, bm, (short)0, acc, false, false);
    }
#pragma unroll
    for (int i = 0; i < 8; ++i) {
      const int r = row0 + mh + i;
      if (r < mValid) {
        if (Cs) Cs[r * DIM + col0 + nl] = (__bf16)acc[i];
        if (Cf) Cf[r * DIM + col0 + nl] = acc[i];
      }
    }
  }
}

// ============================================================================
// Kernel 0: transpose + f32->bf16 convert the six 256x256 weights so the WMMA
// B fragment is a contiguous 32B load.  grid = (256 rows, 6 weights), 256 thr.
// ============================================================================
__global__ __launch_bounds__(256) void prep_weights_kernel(
    const float* __restrict__ w0, const float* __restrict__ w1,
    const float* __restrict__ w2, const float* __restrict__ w3,
    const float* __restrict__ w4, const float* __restrict__ w5,
    __bf16* __restrict__ o0, __bf16* __restrict__ o1,
    __bf16* __restrict__ o2, __bf16* __restrict__ o3,
    __bf16* __restrict__ o4, __bf16* __restrict__ o5)
{
  const float* src; __bf16* dst;
  switch (blockIdx.y) {
    case 0: src = w0; dst = o0; break;
    case 1: src = w1; dst = o1; break;
    case 2: src = w2; dst = o2; break;
    case 3: src = w3; dst = o3; break;
    case 4: src = w4; dst = o4; break;
    default: src = w5; dst = o5; break;
  }
  const int n = blockIdx.x;
  const int k = threadIdx.x;
  dst[(size_t)n * DIM + k] = (__bf16)src[(size_t)k * DIM + n];
}

// ============================================================================
// Kernel 1: fused satellite attention block (32 rows / workgroup, 8 waves).
// ============================================================================
__global__ __launch_bounds__(256) void sat_kernel(
    const float* __restrict__ h, const float* __restrict__ e,
    const float* __restrict__ svec,
    const __bf16* __restrict__ wqt, const __bf16* __restrict__ wkt,
    const __bf16* __restrict__ wvt, const __bf16* __restrict__ wot,
    const float* __restrict__ gvec, const float* __restrict__ bvec,
    float* __restrict__ hout, __bf16* __restrict__ houtbf)
{
  extern __shared__ char smem[];
  __bf16* hs   = (__bf16*)(smem + HS_OFF);
  __bf16* es   = (__bf16*)(smem + ES_OFF);
  __bf16* ss   = (__bf16*)(smem + SS_OFF);
  __bf16* qs   = (__bf16*)(smem + Q_OFF);
  __bf16* hk   = (__bf16*)(smem + HK_OFF);
  __bf16* hv   = (__bf16*)(smem + HV_OFF);
  __bf16* ek   = (__bf16*)(smem + EK_OFF);
  __bf16* ev   = (__bf16*)(smem + EV_OFF);
  __bf16* sk   = (__bf16*)(smem + SK_OFF);
  __bf16* svls = (__bf16*)(smem + SV_OFF);
  __bf16* os   = (__bf16*)(smem + O_OFF);
  float*  opf  = (float*)(smem + OPF_OFF);

  const int tid  = threadIdx.x;
  const int wave = tid >> 5;
  const int lane = tid & 31;
  const int R    = blockIdx.x * BM;     // global flat row base (never crosses batch)
  const int bb   = R / SEQ;
  const int s0   = R % SEQ;

  // ---- stage inputs (f32 -> bf16) ----
  for (int idx = tid; idx < 48 * DIM; idx += 256) {       // h rows with +-1 halo
    const int r = idx >> 8, c = idx & 255;
    const int seq = s0 + r - 1;
    float v = 0.f;
    if (r < BM + 2 && seq >= 0 && seq < SEQ)
      v = h[((size_t)bb * SEQ + seq) * DIM + c];
    hs[idx] = (__bf16)v;
  }
  for (int idx = tid; idx < BM * DIM; idx += 256) {
    const int r = idx >> 8, c = idx & 255;
    es[idx] = (__bf16)e[((size_t)bb * SEQ + s0 + r) * DIM + c];
  }
  for (int idx = tid; idx < 16 * DIM; idx += 256) {
    const int r = idx >> 8, c = idx & 255;
    ss[idx] = (__bf16)((r == 0) ? svec[bb * DIM + c] : 0.f);
  }
  __syncthreads();

  // ---- projections (WMMA) ----
  gemm_bf16(hs + DIM, wqt, qs,   nullptr, 2, BM,     wave, lane);  // q of h rows
  gemm_bf16(hs,       wkt, hk,   nullptr, 3, BM + 2, wave, lane);  // k of halo'd h
  gemm_bf16(hs,       wvt, hv,   nullptr, 3, BM + 2, wave, lane);
  gemm_bf16(es,       wkt, ek,   nullptr, 2, BM,     wave, lane);
  gemm_bf16(es,       wvt, ev,   nullptr, 2, BM,     wave, lane);
  gemm_bf16(ss,       wkt, sk,   nullptr, 1, 1,      wave, lane);
  gemm_bf16(ss,       wvt, svls, nullptr, 1, 1,      wave, lane);
  __syncthreads();

  // ---- 5-way attention per (row, head) ----
  {
    const int row = tid >> 3;      // 0..31
    const int hd  = tid & 7;       // 0..7
    float qv[DK];
#pragma unroll
    for (int d = 0; d < DK; ++d) qv[d] = (float)qs[row * DIM + hd * DK + d];
    const __bf16* kp[5] = { hk + row * DIM, hk + (row + 1) * DIM,
                            hk + (row + 2) * DIM, ek + row * DIM, sk };
    const __bf16* vp[5] = { hv + row * DIM, hv + (row + 1) * DIM,
                            hv + (row + 2) * DIM, ev + row * DIM, svls };
    float sc[5];
    float mx = -1e30f;
#pragma unroll
    for (int c = 0; c < 5; ++c) {
      float d0 = 0.f;
#pragma unroll
      for (int d = 0; d < DK; ++d) d0 += qv[d] * (float)kp[c][hd * DK + d];
      sc[c] = d0 * SCALE;
      mx = fmaxf(mx, sc[c]);
    }
    float l = 0.f;
#pragma unroll
    for (int c = 0; c < 5; ++c) { sc[c] = __expf(sc[c] - mx); l += sc[c]; }
    const float inv = 1.f / l;
#pragma unroll
    for (int d = 0; d < DV; ++d) {
      float o = 0.f;
#pragma unroll
      for (int c = 0; c < 5; ++c) o += sc[c] * (float)vp[c][hd * DV + d];
      os[row * DIM + hd * DV + d] = (__bf16)(o * inv);
    }
  }
  __syncthreads();

  // ---- output projection (WMMA, f32 out) ----
  gemm_bf16(os, wot, nullptr, opf, 2, BM, wave, lane);
  __syncthreads();

  // ---- residual + LayerNorm (wave32 shuffles), write f32 + bf16 copies ----
#pragma unroll
  for (int rr = 0; rr < 4; ++rr) {
    const int row = wave * 4 + rr;
    const size_t grow = (size_t)bb * SEQ + s0 + row;
    float xv[8], sum = 0.f, sq = 0.f;
#pragma unroll
    for (int i = 0; i < 8; ++i) {
      const int j = lane + 32 * i;
      const float x = opf[row * DIM + j] + h[grow * DIM + j];
      xv[i] = x; sum += x; sq += x * x;
    }
    for (int o = 16; o > 0; o >>= 1) {
      sum += __shfl_xor(sum, o, 32);
      sq  += __shfl_xor(sq,  o, 32);
    }
    const float mu   = sum * (1.f / DIM);
    const float var  = sq * (1.f / DIM) - mu * mu;
    const float rstd = rsqrtf(var + 1e-6f);
#pragma unroll
    for (int i = 0; i < 8; ++i) {
      const int j = lane + 32 * i;
      const float y = (xv[i] - mu) * rstd * gvec[j] + bvec[j];
      hout[grow * DIM + j]   = y;
      houtbf[grow * DIM + j] = (__bf16)y;
    }
  }
}

// ============================================================================
// Kernel 2: relay attention — streaming online softmax over S+1 keys.
// One workgroup per batch; wave = head, lane = DV dim.
// Chunks >= 1 are staged by the Tensor Data Mover (tensor_load_to_lds):
// a pure 64x256 bf16 tile copy from the bf16 h_out mirror; tensor_dim1 is set
// to the remaining row count so TDM's OOB-read-returns-zero handles the
// ragged final chunk.  Chunk 0 (s row + index shift) uses the scalar path.
// ============================================================================
__global__ __launch_bounds__(256) void relay_kernel(
    const float* __restrict__ svec, const float* __restrict__ rel_wq,
    const __bf16* __restrict__ rwkt, const __bf16* __restrict__ rwvt,
    const __bf16* __restrict__ houtbf, float* __restrict__ orl)
{
  extern __shared__ char smem[];
  __bf16* sh = (__bf16*)(smem + RSH_OFF);
  __bf16* kr = (__bf16*)(smem + RKR_OFF);
  __bf16* vr = (__bf16*)(smem + RVR_OFF);
  float*  qr = (float*)(smem + RQR_OFF);
  const int tid = threadIdx.x, wave = tid >> 5, lane = tid & 31;
  const int bb  = blockIdx.x;

  // qr = s @ rel_wq  (one output element per thread; tiny matvec)
  {
    float a = 0.f;
    for (int k = 0; k < DIM; ++k)
      a += svec[bb * DIM + k] * rel_wq[(size_t)k * DIM + tid];
    qr[tid] = a;
  }
  __syncthreads();

  const int hd = wave;
  float qh[DK];
#pragma unroll
  for (int d = 0; d < DK; ++d) qh[d] = qr[hd * DK + d];

  float m = -1e30f, l = 0.f, acc = 0.f;
  const int NCH = (SEQ + 64) / 64;      // 65 chunks cover SEQ+1 rows
  for (int c = 0; c < NCH; ++c) {
    if (c >= 1) {
      // ---- TDM staging: rows jg = c*64 .. c*64+63 map to hout rows jg-1 ----
      if (wave == 0) {
        const unsigned long long gaddr = (unsigned long long)(const void*)
            (houtbf + ((size_t)bb * SEQ + (size_t)c * 64 - 1) * DIM);
        const unsigned rows_left = (unsigned)(SEQ + 1 - c * 64); // OOB rows -> 0
        v4u g0;
        g0[0] = 1u;                                  // count=1 (valid user D#)
        g0[1] = (unsigned)RSH_OFF;                   // lds_addr = 0
        g0[2] = (unsigned)(gaddr & 0xFFFFFFFFu);     // global_addr[31:0]
        g0[3] = (unsigned)((gaddr >> 32) & 0x01FFFFFFu) | (2u << 30); // type=2
        v8i_ g1;
        g1[0] = (int)(1u << 16);                     // data_size = 2 bytes
        g1[1] = (int)(((unsigned)DIM & 0xFFFFu) << 16);          // tdim0 lo16
        g1[2] = (int)((((unsigned)DIM >> 16) & 0xFFFFu) |
                      ((rows_left & 0xFFFFu) << 16));            // tdim0 hi/tdim1 lo
        g1[3] = (int)(((rows_left >> 16) & 0xFFFFu) |
                      (((unsigned)DIM & 0xFFFFu) << 16));        // tdim1 hi/tile0=256
        g1[4] = (int)64u;                            // tile_dim1=64, tile_dim2=0
        g1[5] = (int)DIM;                            // tensor_dim0_stride = 256
        g1[6] = 0;
        g1[7] = 0;
        v4i_ gz = {0, 0, 0, 0};
#if __clang_major__ >= 23
        v8i_ gz8 = {0, 0, 0, 0, 0, 0, 0, 0};
        __builtin_amdgcn_tensor_load_to_lds(g0, g1, gz, gz, gz8, 0);
#else
        __builtin_amdgcn_tensor_load_to_lds(g0, g1, gz, gz, 0);
#endif
        __builtin_amdgcn_s_wait_tensorcnt(0);
      }
    } else {
      // ---- scalar staging for chunk 0 (row 0 = s itself, then shift) ----
      for (int idx = tid; idx < 64 * DIM; idx += 256) {
        const int jj = idx >> 8, col = idx & 255;
        const int jg = c * 64 + jj;
        __bf16 v = (__bf16)0.f;
        if (jg == 0)        v = (__bf16)svec[bb * DIM + col];
        else if (jg <= SEQ) v = houtbf[((size_t)bb * SEQ + (jg - 1)) * DIM + col];
        sh[idx] = v;
      }
    }
    __syncthreads();
    gemm_bf16(sh, rwkt, kr, nullptr, 4, 64, wave, lane);
    gemm_bf16(sh, rwvt, vr, nullptr, 4, 64, wave, lane);
    __syncthreads();

    // scores for j = lane and j = lane + 32
    float s0 = -1e30f, s1 = -1e30f;
    if (c * 64 + lane <= SEQ) {
      float d0 = 0.f;
#pragma unroll
      for (int d = 0; d < DK; ++d) d0 += qh[d] * (float)kr[lane * DIM + hd * DK + d];
      s0 = d0 * SCALE;
    }
    if (c * 64 + lane + 32 <= SEQ) {
      float d1 = 0.f;
#pragma unroll
      for (int d = 0; d < DK; ++d) d1 += qh[d] * (float)kr[(lane + 32) * DIM + hd * DK + d];
      s1 = d1 * SCALE;
    }
    float lm = fmaxf(s0, s1);
    for (int o = 16; o > 0; o >>= 1) lm = fmaxf(lm, __shfl_xor(lm, o, 32));
    const float nm = fmaxf(m, lm);
    const float e0 = __expf(s0 - nm), e1 = __expf(s1 - nm);
    float ls = e0 + e1;
    for (int o = 16; o > 0; o >>= 1) ls += __shfl_xor(ls, o, 32);
    const float corr = __expf(m - nm);
    float a2 = 0.f;
    for (int j = 0; j < 32; ++j) {
      const float ej = __shfl(e0, j, 32);
      a2 += ej * (float)vr[j * DIM + hd * DV + lane];
    }
    for (int j = 0; j < 32; ++j) {
      const float ej = __shfl(e1, j, 32);
      a2 += ej * (float)vr[(j + 32) * DIM + hd * DV + lane];
    }
    acc = acc * corr + a2;
    l   = l * corr + ls;
    m   = nm;
    __syncthreads();
  }
  orl[bb * DIM + hd * DV + lane] = acc / l;
}

// ============================================================================
// Kernel 3: s_out = LayerNorm(orl @ rel_wo + s)
// ============================================================================
__global__ __launch_bounds__(256) void finish_kernel(
    const float* __restrict__ orl, const float* __restrict__ svec,
    const float* __restrict__ rel_wo, const float* __restrict__ gvec,
    const float* __restrict__ bvec, float* __restrict__ out_s)
{
  const int bb = blockIdx.x, n = threadIdx.x;
  __shared__ float xs[DIM];
  __shared__ float mu_s, var_s;
  float x = svec[bb * DIM + n];
  for (int k = 0; k < DIM; ++k)
    x += orl[bb * DIM + k] * rel_wo[(size_t)k * DIM + n];
  xs[n] = x;
  __syncthreads();
  if (n < 32) {
    float sum = 0.f, sq = 0.f;
    for (int i = n; i < DIM; i += 32) { const float v = xs[i]; sum += v; sq += v * v; }
    for (int o = 16; o > 0; o >>= 1) {
      sum += __shfl_xor(sum, o, 32);
      sq  += __shfl_xor(sq,  o, 32);
    }
    if (n == 0) {
      const float mu = sum * (1.f / DIM);
      mu_s  = mu;
      var_s = sq * (1.f / DIM) - mu * mu;
    }
  }
  __syncthreads();
  out_s[bb * DIM + n] = (x - mu_s) * rsqrtf(var_s + 1e-6f) * gvec[n] + bvec[n];
}

// ============================================================================
extern "C" void kernel_launch(void* const* d_in, const int* in_sizes, int n_in,
                              void* d_out, int out_size, void* d_ws, size_t ws_size,
                              hipStream_t stream) {
  (void)in_sizes; (void)n_in; (void)out_size; (void)ws_size;
  const float* h      = (const float*)d_in[0];
  const float* e      = (const float*)d_in[1];
  const float* s      = (const float*)d_in[2];
  const float* sat_wq = (const float*)d_in[3];
  const float* sat_wk = (const float*)d_in[4];
  const float* sat_wv = (const float*)d_in[5];
  const float* sat_wo = (const float*)d_in[6];
  const float* sat_g  = (const float*)d_in[7];
  const float* sat_b  = (const float*)d_in[8];
  const float* rel_wq = (const float*)d_in[9];
  const float* rel_wk = (const float*)d_in[10];
  const float* rel_wv = (const float*)d_in[11];
  const float* rel_wo = (const float*)d_in[12];
  const float* rel_g  = (const float*)d_in[13];
  const float* rel_b  = (const float*)d_in[14];

  // workspace layout
  const size_t WB = (size_t)DIM * DIM * sizeof(__bf16);   // 131072 B per weight
  char* ws = (char*)d_ws;
  __bf16* wqt    = (__bf16*)(ws + 0 * WB);
  __bf16* wkt    = (__bf16*)(ws + 1 * WB);
  __bf16* wvt    = (__bf16*)(ws + 2 * WB);
  __bf16* wot    = (__bf16*)(ws + 3 * WB);
  __bf16* rwkt   = (__bf16*)(ws + 4 * WB);
  __bf16* rwvt   = (__bf16*)(ws + 5 * WB);
  __bf16* houtbf = (__bf16*)(ws + 6 * WB);
  float*  orl    = (float*)(ws + 6 * WB + (size_t)NB * SEQ * DIM * sizeof(__bf16));

  float* hout  = (float*)d_out;                          // [B,S,D]
  float* out_s = (float*)d_out + (size_t)NB * SEQ * DIM; // [B,D]

  prep_weights_kernel<<<dim3(DIM, 6), 256, 0, stream>>>(
      sat_wq, sat_wk, sat_wv, sat_wo, rel_wk, rel_wv,
      wqt, wkt, wvt, wot, rwkt, rwvt);

  sat_kernel<<<dim3(NB * SEQ / BM), 256, SAT_LDS, stream>>>(
      h, e, s, wqt, wkt, wvt, wot, sat_g, sat_b, hout, houtbf);

  relay_kernel<<<dim3(NB), 256, REL_LDS, stream>>>(
      s, rel_wq, rwkt, rwvt, houtbf, orl);

  finish_kernel<<<dim3(NB), 256, 0, stream>>>(
      orl, s, rel_wo, rel_g, rel_b, out_s);
}
